// VQVAE256D_61907658605312
// MI455X (gfx1250) — matrix operand order repaired and checked by reading, not compiled, not run on hardware
//
#include <hip/hip_runtime.h>

// ---------------------------------------------------------------------------
// Types for CDNA5 WMMA (wave32): v_wmma_f32_16x16x32_bf16
// ---------------------------------------------------------------------------
typedef __attribute__((ext_vector_type(16))) __bf16 v16bf;
typedef __attribute__((ext_vector_type(8)))  __bf16 v8bf;
typedef __attribute__((ext_vector_type(8)))  float  v8f;

// Async Global->LDS (CDNA5): GLOBAL_LOAD_ASYNC_TO_LDS_B128 + S_WAIT_ASYNCCNT.
// Probe-derived signature: (int4 addrspace(1)*, int4 addrspace(3)*, imm, imm).
// Gated on builtin availability; falls back to global_load_b128/ds_store_b128.
#if __has_builtin(__builtin_amdgcn_global_load_async_to_lds_b128) && \
    __has_builtin(__builtin_amdgcn_s_wait_asynccnt)
#define USE_ASYNC_LDS 1
#else
#define USE_ASYNC_LDS 0
#endif

typedef int v4i_vs __attribute__((vector_size(16)));
typedef __attribute__((address_space(1))) v4i_vs* gbl4_t;
typedef __attribute__((address_space(3))) v4i_vs* lds4_t;

__device__ __forceinline__ void copy16_g2lds(void* lds, const void* g) {
#if USE_ASYNC_LDS
  __builtin_amdgcn_global_load_async_to_lds_b128((gbl4_t)g, (lds4_t)lds, 0, 0);
#else
  *(uint4*)lds = *(const uint4*)g;
#endif
}

__device__ __forceinline__ void async_wait() {
#if USE_ASYNC_LDS
  __builtin_amdgcn_s_wait_asynccnt(0);
#endif
}

__device__ __forceinline__ unsigned short f2bf(float f) {
  union { float f; unsigned u; } v; v.f = f;
  unsigned u = v.u;
  return (unsigned short)((u + 0x7FFFu + ((u >> 16) & 1u)) >> 16); // RNE
}

__device__ __forceinline__ v8f v8f_zero() {
  v8f z;
#pragma unroll
  for (int i = 0; i < 8; ++i) z[i] = 0.0f;
  return z;
}

__device__ __forceinline__ v16bf frag_combine(v8bf lo, v8bf hi) {
  v16bf f;
#pragma unroll
  for (int i = 0; i < 8; ++i) { f[i] = lo[i]; f[i + 8] = hi[i]; }
  return f;
}

// ---------------------------------------------------------------------------
// conv1: 3->64, k4 s2 p1 + ReLU, direct VALU (only 1.6 GFLOP, input L2-resident)
// grid covers exactly B*64*32*32 = 16,777,216 threads; i is the NCHW linear idx
// ---------------------------------------------------------------------------
__global__ __launch_bounds__(256) void conv1_kernel(
    const float* __restrict__ x, const float* __restrict__ w,
    const float* __restrict__ bias, float* __restrict__ out) {
  const int i  = blockIdx.x * 256 + threadIdx.x;
  const int ow = i & 31, oh = (i >> 5) & 31, co = (i >> 10) & 63, b = i >> 16;
  __shared__ float sw[48];
  __shared__ float sb;
  if (threadIdx.x < 48) sw[threadIdx.x] = w[co * 48 + threadIdx.x];
  if (threadIdx.x == 0) sb = bias[co];
  __syncthreads();
  float acc = sb;
  const float* xb = x + (size_t)b * 3 * 64 * 64;
#pragma unroll
  for (int ci = 0; ci < 3; ++ci) {
#pragma unroll
    for (int kh = 0; kh < 4; ++kh) {
      const int ih = oh * 2 - 1 + kh;
      if ((unsigned)ih < 64u) {
#pragma unroll
        for (int kw = 0; kw < 4; ++kw) {
          const int iw = ow * 2 - 1 + kw;
          if ((unsigned)iw < 64u)
            acc = fmaf(xb[(ci * 64 + ih) * 64 + iw], sw[(ci * 4 + kh) * 4 + kw], acc);
        }
      }
    }
  }
  out[i] = fmaxf(acc, 0.0f);
}

// ---------------------------------------------------------------------------
// Weight packing: fp32 -> bf16, same layout ([N][K] row-major already)
// ---------------------------------------------------------------------------
__global__ void pack_bf16(const float* __restrict__ src,
                          unsigned short* __restrict__ dst, int n) {
  int i = blockIdx.x * 256 + threadIdx.x;
  if (i < n) dst[i] = f2bf(src[i]);
}

// Deconv weights: w[Cin][Cout][4][4] -> dst[class][Cout][Cin][tap] bf16,
// tap = ty*2+tx, kh = (1-py)+2*ty, kw = (1-px)+2*tx  (class = py*2+px)
__global__ void pack_decw(const float* __restrict__ w,
                          unsigned short* __restrict__ dst,
                          int Cin, int Cout, int total) {
  int i = blockIdx.x * 256 + threadIdx.x;
  if (i >= total) return;
  int t = i & 3;  int r = i >> 2;
  int ci = r % Cin; r /= Cin;
  int co = r % Cout; int p = r / Cout;
  int py = p >> 1, px = p & 1, ty = t >> 1, tx = t & 1;
  int kh = (1 - py) + 2 * ty, kw = (1 - px) + 2 * tx;
  float v = w[(((size_t)ci * Cout + co) * 4 + kh) * 4 + kw];
  dst[(((size_t)p * Cout + co) * Cin + ci) * 4 + t] = f2bf(v);
}

// VQ prep: per-code norms ||e_k||^2, zero histogram + sqsum
__global__ void vq_prep(const float* __restrict__ cb, float* __restrict__ cnorm,
                        unsigned* __restrict__ hist, float* __restrict__ sq) {
  int t = threadIdx.x;
  float s = 0.f;
  for (int d = 0; d < 256; ++d) { float v = cb[t * 256 + d]; s = fmaf(v, v, s); }
  cnorm[t] = s;
  hist[t] = 0u;
  if (t == 0) *sq = 0.f;
}

// ---------------------------------------------------------------------------
// Generic implicit-GEMM WMMA kernel (bf16 A/B, f32 accum).
//   MODE 0: conv k4 s2 p1 (im2col gather)        rows m = (b, oh, ow)
//   MODE 1: 1x1 conv                             rows m = (b, oh, ow)
//   MODE 2: transposed conv k4 s2 p1, parity     rows m = (b, ohh, oww),
//           class = blockIdx.z, oh = 2*ohh+py
//   EPI 0: bias + ReLU -> NCHW fp32
//   EPI 1: bias -> fp32 encoded + bf16 copy (for VQ A-matrix)
//   EPI 2: bias + ReLU -> LDS, fused 1x1 (64->3) + sigmoid -> decoded
// 256 threads = 8 waves; tile BMxBN, BK=32; double-buffered LDS, 1 barrier/step
// B-tile staging uses GLOBAL_LOAD_ASYNC_TO_LDS_B128 when available.
// ---------------------------------------------------------------------------
template <int MODE, int EPI, int BM, int BN, int WR, int WC,
          int CIN, int IH, int IW, int KTOT>
__global__ __launch_bounds__(256) void gemm_wmma(
    const float* __restrict__ Ain,                 // fp32 activations
    const unsigned short* __restrict__ Bpack,      // bf16 [class][N][KTOT]
    const float* __restrict__ bias,
    float* __restrict__ Out,
    unsigned short* __restrict__ OutB,             // EPI 1 bf16 copy
    const float* __restrict__ W3,                  // EPI 2: dw3 [64][3]
    const float* __restrict__ B3,                  // EPI 2: db3 [3]
    int NCH) {
  static_assert(WR * WC == 8, "8 waves");
  constexpr int WTM = BM / WR, WTN = BN / WC;
  constexpr int MI = WTM / 16, NI = WTN / 16;
  constexpr int LD = 40;                            // 32 + 8 pad (ushort units)
  constexpr int KSTEPS = KTOT / 32;
  constexpr int OHW = (MODE == 0) ? IW / 2 : IW;    // m-space row width
  constexpr int RPI = ((MODE == 0) ? IH / 2 : IH) * OHW;

  __shared__ alignas(16) unsigned short sA[2][BM * LD];
  __shared__ alignas(16) unsigned short sB[2][BN * LD];
  __shared__ float sOut[(EPI == 2) ? (BM * (BN + 1)) : 1];
  __shared__ float sW3[(EPI == 2) ? 195 : 1];

  const int tid = threadIdx.x;
  const int wave = tid >> 5, lane = tid & 31;
  const int lr = lane & 15, hi = lane >> 4;
  const int wr = wave / WC, wc = wave % WC;
  const int m0 = blockIdx.x * BM;
  const int n0 = blockIdx.y * BN;
  const int cls = blockIdx.z;
  const int py = cls >> 1, px = cls & 1;

  const unsigned short* Bp = Bpack + (size_t)cls * (size_t)NCH * KTOT;

  v8f acc[MI][NI];
#pragma unroll
  for (int i = 0; i < MI; ++i)
#pragma unroll
    for (int j = 0; j < NI; ++j) acc[i][j] = v8f_zero();

  // ---- staging: im2col-gather A (fp32 -> bf16) into fragment-friendly LDS ----
  auto stageA = [&](int buf, int k0) {
    constexpr int ELEMS = BM * 32 / 256;            // per-thread elements
#pragma unroll
    for (int e = 0; e < ELEMS; ++e) {
      const int linear = tid * ELEMS + e;
      const int row = linear >> 5, col = linear & 31;
      const int gm = m0 + row;
      const int k = k0 + col;
      float v = 0.0f;
      if constexpr (MODE == 0) {
        const int b = gm / RPI, r = gm % RPI;
        const int oh = r / OHW, ow = r % OHW;
        const int ci = k >> 4, kh = (k >> 2) & 3, kw = k & 3;
        const int ihh = oh * 2 - 1 + kh, iww = ow * 2 - 1 + kw;
        if ((unsigned)ihh < (unsigned)IH && (unsigned)iww < (unsigned)IW)
          v = Ain[(((size_t)b * CIN + ci) * IH + ihh) * IW + iww];
      } else if constexpr (MODE == 1) {
        const int b = gm / RPI, r = gm % RPI;
        v = Ain[(((size_t)b * CIN + k) * RPI) + r];
      } else {
        const int b = gm / RPI, r = gm % RPI;
        const int ohh = r / OHW, oww = r % OHW;
        const int ci = k >> 2, ty = (k >> 1) & 1, tx = k & 1;
        const int ihh = ohh + py - ty, iww = oww + px - tx;
        if ((unsigned)ihh < (unsigned)IH && (unsigned)iww < (unsigned)IW)
          v = Ain[(((size_t)b * CIN + ci) * IH + ihh) * IW + iww];
      }
      sA[buf][row * LD + col] = f2bf(v);
    }
  };
  // ---- staging: pre-packed bf16 weights, async b128 global->LDS ----
  auto stageB = [&](int buf, int k0) {
    constexpr int CH = (BN * 4) / 256;              // 16B chunks per thread
#pragma unroll
    for (int j = 0; j < CH; ++j) {
      const int c = tid + 256 * j;
      const int nrow = c >> 2, q = c & 3;
      copy16_g2lds(&sB[buf][nrow * LD + q * 8],
                   Bp + (size_t)(n0 + nrow) * KTOT + k0 + q * 8);
    }
  };

  stageA(0, 0);
  stageB(0, 0);
  async_wait();
  __syncthreads();

  for (int s = 0; s < KSTEPS; ++s) {
    const int cur = s & 1;
    if (s + 1 < KSTEPS) {                           // overlap next stage w/ math
      stageA(cur ^ 1, (s + 1) * 32);
      stageB(cur ^ 1, (s + 1) * 32);
    }
    if (s + 2 < KSTEPS)                             // global_prefetch_b8
      __builtin_prefetch((const void*)(Bp + (size_t)(n0 + (tid & (BN - 1))) * KTOT
                                       + (s + 2) * 32), 0, 0);

    v16bf aF[MI];
#pragma unroll
    for (int mi = 0; mi < MI; ++mi) {
      const unsigned short* p = &sA[cur][(wr * WTM + mi * 16 + lr) * LD + hi * 8];
      aF[mi] = frag_combine(*(const v8bf*)(const void*)p,
                            *(const v8bf*)(const void*)(p + 16));
    }
    v16bf bF[NI];
#pragma unroll
    for (int ni = 0; ni < NI; ++ni) {
      const unsigned short* p = &sB[cur][(wc * WTN + ni * 16 + lr) * LD + hi * 8];
      bF[ni] = frag_combine(*(const v8bf*)(const void*)p,
                            *(const v8bf*)(const void*)(p + 16));
    }
#pragma unroll
    for (int mi = 0; mi < MI; ++mi)
#pragma unroll
      for (int ni = 0; ni < NI; ++ni)
        acc[mi][ni] = __builtin_amdgcn_wmma_f32_16x16x32_bf16(
            false, aF[mi], false, bF[ni], (short)0, acc[mi][ni], false, false);
    async_wait();                                   // next tile landed in LDS
    __syncthreads();
  }

  // ----------------------------- epilogue ----------------------------------
#pragma unroll
  for (int mi = 0; mi < MI; ++mi) {
#pragma unroll
    for (int ni = 0; ni < NI; ++ni) {
      const int ncol = wc * WTN + ni * 16 + lr;
      const int col = n0 + ncol;
      const float bv = bias[col];
#pragma unroll
      for (int j = 0; j < 8; ++j) {
        const int lrow = wr * WTM + mi * 16 + j + 8 * hi;
        const int gm = m0 + lrow;
        float v = acc[mi][ni][j] + bv;
        if constexpr (EPI == 0) {
          v = fmaxf(v, 0.0f);
          const int b = gm / RPI, r = gm % RPI;
          if constexpr (MODE == 2) {
            const int ohh = r / OHW, oww = r % OHW;
            const int oh = 2 * ohh + py, ow = 2 * oww + px;
            Out[(((size_t)b * NCH + col) * (2 * IH) + oh) * (2 * IW) + ow] = v;
          } else {
            const int oh = r / OHW, ow = r % OHW;
            constexpr int OH = (MODE == 0) ? IH / 2 : IH;
            constexpr int OW = (MODE == 0) ? IW / 2 : IW;
            Out[(((size_t)b * NCH + col) * OH + oh) * OW + ow] = v;
          }
        } else if constexpr (EPI == 1) {
          const int b = gm / RPI, r = gm % RPI;
          const size_t idx = ((size_t)b * NCH + col) * RPI + r;
          Out[idx] = v;
          OutB[idx] = f2bf(v);                       // bf16 A-matrix for VQ
        } else {
          sOut[lrow * (BN + 1) + ncol] = fmaxf(v, 0.0f);
        }
      }
    }
  }

  if constexpr (EPI == 2) {                          // fused 1x1 (64->3)+sigmoid
    if (tid < 195) sW3[tid] = (tid < 192) ? W3[tid] : B3[tid - 192];
    __syncthreads();
    if (tid < BM) {
      const int lrow = tid, gm = m0 + lrow;
      const int b = gm / RPI, r = gm % RPI;
      const int ohh = r / OHW, oww = r % OHW;
      const int oh = 2 * ohh + py, ow = 2 * oww + px;
      float s0 = sW3[192], s1 = sW3[193], s2 = sW3[194];
#pragma unroll
      for (int ci = 0; ci < BN; ++ci) {
        const float a = sOut[lrow * (BN + 1) + ci];
        s0 = fmaf(a, sW3[ci * 3 + 0], s0);
        s1 = fmaf(a, sW3[ci * 3 + 1], s1);
        s2 = fmaf(a, sW3[ci * 3 + 2], s2);
      }
      const size_t hw = (size_t)(2 * IH) * (2 * IW);
      const size_t sp = (size_t)oh * (2 * IW) + ow;
      float* o = Out + (size_t)b * 3 * hw + sp;
      o[0 * hw] = 1.0f / (1.0f + __expf(-s0));
      o[1 * hw] = 1.0f / (1.0f + __expf(-s1));
      o[2 * hw] = 1.0f / (1.0f + __expf(-s2));
    }
  }
}

// ---------------------------------------------------------------------------
// VQ: scores s[n,k] = ||e_k||^2 - 2 * <flat_n, e_k> via WMMA (all 256 codes per
// block -> complete argmin in LDS), then fused gather + MSE + histogram.
// 32 rows/block, 8 waves as 2x4 (wave tile 16x64), K = 256 in 8 steps.
// ---------------------------------------------------------------------------
__global__ __launch_bounds__(256) void vq_kernel(
    const unsigned short* __restrict__ encB, const unsigned short* __restrict__ Bq,
    const float* __restrict__ cnorm, const float* __restrict__ encF,
    const float* __restrict__ cbF, float* __restrict__ quant,
    unsigned* __restrict__ hist, float* __restrict__ sqsum) {
  constexpr int LD = 40;
  __shared__ alignas(16) unsigned short sA[32 * LD];
  __shared__ alignas(16) unsigned short sB[256 * LD];
  __shared__ float sCn[256];
  __shared__ float sScore[32 * 260];
  __shared__ float sRedMin[256];
  __shared__ int sRedIdx[256];
  __shared__ int sIdx[32];
  __shared__ float sLs[256];

  const int tid = threadIdx.x;
  const int wave = tid >> 5, lane = tid & 31;
  const int lr = lane & 15, hi = lane >> 4;
  const int wr = wave >> 2, wc = wave & 3;           // 2 x 4 wave grid
  const int m0 = blockIdx.x * 32;

  sCn[tid] = cnorm[tid];
  v8f acc[4];
#pragma unroll
  for (int i = 0; i < 4; ++i) acc[i] = v8f_zero();

  for (int s = 0; s < 8; ++s) {
    const int k0 = s * 32;
    if (tid < 128) {                                  // A: 32 x 32 bf16, b128
      const int row = tid >> 2, q = tid & 3;
      copy16_g2lds(&sA[row * LD + q * 8],
                   encB + (size_t)(m0 + row) * 256 + k0 + q * 8);
    }
#pragma unroll
    for (int j = 0; j < 4; ++j) {                     // B: 256 x 32 bf16
      const int c = tid + 256 * j;
      const int row = c >> 2, q = c & 3;
      copy16_g2lds(&sB[row * LD + q * 8],
                   Bq + (size_t)row * 256 + k0 + q * 8);
    }
    async_wait();
    __syncthreads();
    v16bf aF;
    {
      const unsigned short* p = &sA[(wr * 16 + lr) * LD + hi * 8];
      aF = frag_combine(*(const v8bf*)(const void*)p,
                        *(const v8bf*)(const void*)(p + 16));
    }
#pragma unroll
    for (int ni = 0; ni < 4; ++ni) {
      const unsigned short* p = &sB[(wc * 64 + ni * 16 + lr) * LD + hi * 8];
      v16bf bF = frag_combine(*(const v8bf*)(const void*)p,
                              *(const v8bf*)(const void*)(p + 16));
      acc[ni] = __builtin_amdgcn_wmma_f32_16x16x32_bf16(
          false, aF, false, bF, (short)0, acc[ni], false, false);
    }
    __syncthreads();
  }

#pragma unroll
  for (int ni = 0; ni < 4; ++ni) {
    const int col = wc * 64 + ni * 16 + lr;
#pragma unroll
    for (int j = 0; j < 8; ++j) {
      const int rl = wr * 16 + j + 8 * hi;
      sScore[rl * 260 + col] = sCn[col] - 2.0f * acc[ni][j];
    }
  }
  __syncthreads();
  {                                                   // per-row argmin, stage 1
    const int r = tid & 31, q = tid >> 5;
    float best = 3.4e38f; int bi = 0;
    for (int c0 = 0; c0 < 32; ++c0) {
      const int c = q * 32 + c0;
      const float v = sScore[r * 260 + c];
      if (v < best) { best = v; bi = c; }
    }
    sRedMin[q * 32 + r] = best;
    sRedIdx[q * 32 + r] = bi;
  }
  __syncthreads();
  if (tid < 32) {                                     // stage 2 + histogram
    float best = sRedMin[tid]; int bi = sRedIdx[tid];
    for (int q = 1; q < 8; ++q) {
      const float v = sRedMin[q * 32 + tid];
      const int ii = sRedIdx[q * 32 + tid];
      if (v < best || (v == best && ii < bi)) { best = v; bi = ii; }
    }
    sIdx[tid] = bi;
    atomicAdd(&hist[bi], 1u);
  }
  __syncthreads();
  float ls = 0.0f;                                    // gather + MSE (fp32)
  for (int i = tid; i < 32 * 256; i += 256) {
    const int r = i >> 8, c = i & 255;
    const int code = sIdx[r];
    const float q = cbF[(size_t)code * 256 + c];
    const size_t g = (size_t)(m0 + r) * 256 + c;
    const float d = q - encF[g];
    quant[g] = q;
    ls = fmaf(d, d, ls);
  }
  sLs[tid] = ls;
  __syncthreads();
  for (int s2 = 128; s2 > 0; s2 >>= 1) {
    if (tid < s2) sLs[tid] += sLs[tid + s2];
    __syncthreads();
  }
  if (tid == 0) atomicAdd(sqsum, sLs[0]);
}

__global__ void vq_finalize(const unsigned* __restrict__ hist,
                            const float* __restrict__ sq,
                            float* __restrict__ outp) {
  __shared__ float red[256];
  const int t = threadIdx.x;
  const float p = (float)hist[t] * (1.0f / 65536.0f);
  red[t] = p * logf(p + 1e-10f);
  __syncthreads();
  for (int s = 128; s > 0; s >>= 1) {
    if (t < s) red[t] += red[t + s];
    __syncthreads();
  }
  if (t == 0) {
    outp[0] = 1.25f * (*sq) * (1.0f / 16777216.0f);   // (1+0.25)*MSE
    outp[1] = __expf(-red[0]);                        // perplexity
  }
}

// ---------------------------------------------------------------------------
extern "C" void kernel_launch(void* const* d_in, const int* in_sizes, int n_in,
                              void* d_out, int out_size, void* d_ws, size_t ws_size,
                              hipStream_t stream) {
  (void)in_sizes; (void)n_in; (void)out_size; (void)ws_size;
  const float* x   = (const float*)d_in[0];
  const float* ew1 = (const float*)d_in[1];
  const float* eb1 = (const float*)d_in[2];
  const float* ew2 = (const float*)d_in[3];
  const float* eb2 = (const float*)d_in[4];
  const float* ew3 = (const float*)d_in[5];
  const float* eb3 = (const float*)d_in[6];
  const float* cb  = (const float*)d_in[7];
  const float* dw1 = (const float*)d_in[8];
  const float* db1 = (const float*)d_in[9];
  const float* dw2 = (const float*)d_in[10];
  const float* db2 = (const float*)d_in[11];
  const float* dw3 = (const float*)d_in[12];
  const float* db3 = (const float*)d_in[13];

  float* out   = (float*)d_out;
  float* dec   = out;                                 // [256,3,64,64]
  float* enc   = out + 3145728;                       // [256,256,16,16]
  float* quant = enc + 16777216;                      // [256,256,16,16]
  float* lossp = quant + 16777216;                    // loss, perplexity

  char* ws = (char*)d_ws;
  size_t off = 0;
  auto alloc = [&](size_t bytes) -> char* {
    char* p = ws + off;
    off += (bytes + 255) & ~(size_t)255;
    return p;
  };
  float* h1            = (float*)alloc((size_t)256 * 64 * 32 * 32 * 4);   // 64MB
  float* h2            = (float*)alloc((size_t)256 * 128 * 16 * 16 * 4);  // 32MB
  float* d1            = (float*)alloc((size_t)256 * 128 * 32 * 32 * 4);  // 128MB
  unsigned short* encB = (unsigned short*)alloc((size_t)65536 * 256 * 2); // 32MB
  unsigned short* B2   = (unsigned short*)alloc((size_t)128 * 1024 * 2);
  unsigned short* B3w  = (unsigned short*)alloc((size_t)256 * 128 * 2);
  unsigned short* Bq   = (unsigned short*)alloc((size_t)256 * 256 * 2);
  unsigned short* Bd1  = (unsigned short*)alloc((size_t)4 * 128 * 1024 * 2);
  unsigned short* Bd2  = (unsigned short*)alloc((size_t)4 * 64 * 512 * 2);
  float* cnorm         = (float*)alloc(256 * 4);
  unsigned* hist       = (unsigned*)alloc(256 * 4);
  float* sqsum         = (float*)alloc(256);

  // weight packing (fp32 -> bf16 GEMM layouts)
  pack_bf16<<<512, 256, 0, stream>>>(ew2, B2, 131072);
  pack_bf16<<<128, 256, 0, stream>>>(ew3, B3w, 32768);
  pack_bf16<<<256, 256, 0, stream>>>(cb, Bq, 65536);
  pack_decw<<<(4 * 256 * 128 * 4 + 255) / 256, 256, 0, stream>>>(dw1, Bd1, 256, 128, 4 * 256 * 128 * 4);
  pack_decw<<<(4 * 128 * 64 * 4 + 255) / 256, 256, 0, stream>>>(dw2, Bd2, 128, 64, 4 * 128 * 64 * 4);
  vq_prep<<<1, 256, 0, stream>>>(cb, cnorm, hist, sqsum);

  // encoder
  conv1_kernel<<<65536, 256, 0, stream>>>(x, ew1, eb1, h1);
  gemm_wmma<0, 0, 128, 128, 4, 2, 64, 32, 32, 1024>
      <<<dim3(512, 1, 1), 256, 0, stream>>>(h1, B2, eb2, h2, nullptr, nullptr, nullptr, 128);
  gemm_wmma<1, 1, 128, 128, 4, 2, 128, 16, 16, 128>
      <<<dim3(512, 2, 1), 256, 0, stream>>>(h2, B3w, eb3, enc, encB, nullptr, nullptr, 256);

  // vector quantization (GEMM + argmin + gather + stats fused)
  vq_kernel<<<2048, 256, 0, stream>>>(encB, Bq, cnorm, enc, cb, quant, hist, sqsum);
  vq_finalize<<<1, 256, 0, stream>>>(hist, sqsum, lossp);

  // decoder: transposed convs as 4 parity-class GEMMs; last layer fused
  gemm_wmma<2, 0, 128, 128, 4, 2, 256, 16, 16, 1024>
      <<<dim3(512, 1, 4), 256, 0, stream>>>(quant, Bd1, db1, d1, nullptr, nullptr, nullptr, 128);
  gemm_wmma<2, 2, 128, 64, 8, 1, 128, 32, 32, 512>
      <<<dim3(2048, 1, 4), 256, 0, stream>>>(d1, Bd2, db2, dec, nullptr, dw3, db3, 64);
}
